// DistanceAwareGNNLayer_1082331758595
// MI455X (gfx1250) — compile-verified, matrix operand order checked by reading, not run on hardware
//
#include <hip/hip_runtime.h>
#include <hip/hip_bf16.h>
#include <math.h>

// Problem constants (from reference): B=8, N=256, D=128, OUT=128, in2=257
#define BATCH 8
#define NN    256
#define DD    128
#define OUTD  128
#define BN    (BATCH * NN)   // 2048

typedef __attribute__((ext_vector_type(16))) _Float16 v16h;
typedef __attribute__((ext_vector_type(8)))  float    v8f;

// ---------------------------------------------------------------------------
// WMMA fragment loaders (wave32, v_wmma_f32_16x16x32_f16).
// A (16x32 f16) per-lane layout (ISA 7.12.2): lane L holds row M=L&15;
//   halves 0..7  -> K = kBase + (L>>4)*8 + {0..7}
//   halves 8..15 -> K = kBase + 16 + (L>>4)*8 + {0..7}
// B (32x16) loaded symmetrically: lane L holds column N=L&15, same K pattern.
// C/D (16x16 f32, 8 VGPRs): lane L -> N=L&15, VGPR r -> M = r + 8*(L>>4).
// ---------------------------------------------------------------------------
__device__ inline v16h load_frag_a(const float* __restrict__ src, int lda,
                                   int mBase, int kBase, int lane) {
  const float* p = src + (size_t)(mBase + (lane & 15)) * lda;
  int kb = (lane >> 4) * 8;
  v16h a;
#pragma unroll
  for (int v = 0; v < 4; ++v) {
    int k0 = kBase + kb + 2 * v;
    a[2 * v]     = (_Float16)p[k0];
    a[2 * v + 1] = (_Float16)p[k0 + 1];
    a[8 + 2 * v]     = (_Float16)p[k0 + 16];
    a[8 + 2 * v + 1] = (_Float16)p[k0 + 17];
  }
  return a;
}

__device__ inline v16h load_frag_b(const float* __restrict__ W, int ldw,
                                   int kBase, int nBase, int lane) {
  int col = nBase + (lane & 15);
  int kb = (lane >> 4) * 8;
  v16h b;
#pragma unroll
  for (int v = 0; v < 4; ++v) {
    int k0 = kBase + kb + 2 * v;
    b[2 * v]     = (_Float16)W[(size_t)k0 * ldw + col];
    b[2 * v + 1] = (_Float16)W[(size_t)(k0 + 1) * ldw + col];
    b[8 + 2 * v]     = (_Float16)W[(size_t)(k0 + 16) * ldw + col];
    b[8 + 2 * v + 1] = (_Float16)W[(size_t)(k0 + 17) * ldw + col];
  }
  return b;
}

// ---------------------------------------------------------------------------
// Kernel 1: HA = h @ Wm1[0:128,:], HB = h @ Wm1[128:256,:]  (WMMA GEMM)
// grid (BN/16, OUT/16, 2), block 32 (one full wave per 16x16 C tile).
// ---------------------------------------------------------------------------
__global__ void k_gemm_h_wm1(const float* __restrict__ h,
                             const float* __restrict__ Wm1,
                             float* __restrict__ HA, float* __restrict__ HB) {
  int lane  = threadIdx.x;
  int mBase = blockIdx.x * 16;
  int nBase = blockIdx.y * 16;
  const float* W = Wm1 + (blockIdx.z ? (size_t)DD * OUTD : 0);
  float* C = blockIdx.z ? HB : HA;
  v8f acc = {};
#pragma unroll
  for (int k = 0; k < DD; k += 32) {
    v16h a = load_frag_a(h, DD, mBase, k, lane);
    v16h b = load_frag_b(W, OUTD, k, nBase, lane);
    acc = __builtin_amdgcn_wmma_f32_16x16x32_f16(false, a, false, b,
                                                 (short)0, acc, false, false);
  }
  int n = nBase + (lane & 15);
  int mAdd = (lane >> 4) * 8;
#pragma unroll
  for (int r = 0; r < 8; ++r)
    C[(size_t)(mBase + r + mAdd) * OUTD + n] = acc[r];
}

// ---------------------------------------------------------------------------
// Kernel 2: per-node attention scalars Ai = h@Wa[0:128], Aj = h@Wa[128:256]
// ---------------------------------------------------------------------------
__global__ void k_attn_scalars(const float* __restrict__ h,
                               const float* __restrict__ Wa,
                               float* __restrict__ Ai, float* __restrict__ Aj) {
  __shared__ float s0[DD], s1[DD];
  int row = blockIdx.x;
  int t = threadIdx.x;
  float hv = h[(size_t)row * DD + t];
  s0[t] = hv * Wa[t];
  s1[t] = hv * Wa[DD + t];
  __syncthreads();
  for (int s = DD / 2; s > 0; s >>= 1) {
    if (t < s) { s0[t] += s0[t + s]; s1[t] += s1[t + s]; }
    __syncthreads();
  }
  if (t == 0) { Ai[row] = s0[0]; Aj[row] = s1[0]; }
}

// ---------------------------------------------------------------------------
// Kernel 3: Wfused = Wm2 @ Wu[128:256,:]; cvec = bm2 @ Wu[128:256,:]
// grid 129 x 128 threads (block 128 computes cvec).  Tiny (2 MFLOP).
// ---------------------------------------------------------------------------
__global__ void k_fuse(const float* __restrict__ Wm2,
                       const float* __restrict__ Wu,
                       const float* __restrict__ bm2,
                       float* __restrict__ Wfused, float* __restrict__ cvec) {
  int k = threadIdx.x;
  int m = blockIdx.x;
  if (m < OUTD) {
    float acc = 0.f;
    for (int t = 0; t < OUTD; ++t)
      acc += Wm2[(size_t)m * OUTD + t] * Wu[(size_t)(DD + t) * OUTD + k];
    Wfused[(size_t)m * OUTD + k] = acc;
  } else {
    float acc = 0.f;
    for (int t = 0; t < OUTD; ++t)
      acc += bm2[t] * Wu[(size_t)(DD + t) * OUTD + k];
    cvec[k] = acc;
  }
}

// ---------------------------------------------------------------------------
// Kernel 4 (dominant): for each (b,i): masked leaky-relu logits, softmax over
// j, then S[b,i,k] = sum_j w_ij * relu(HA_ik + HB_jk + d_ij*wm1d_k + bm1_k),
// wsum[b,i] = sum_j w_ij.  One block of 128 threads (k) per (b,i).
// ---------------------------------------------------------------------------
__global__ void k_softmax_agg(const float* __restrict__ dist,
                              const int*   __restrict__ adj,
                              const float* __restrict__ Ai,
                              const float* __restrict__ Aj,
                              const float* __restrict__ HA,
                              const float* __restrict__ HB,
                              const float* __restrict__ Wm1,
                              const float* __restrict__ bm1,
                              const float* __restrict__ Wa,
                              const float* __restrict__ ba,
                              float* __restrict__ S,
                              float* __restrict__ Wsum) {
  __shared__ float se[NN], sw[NN], sd[NN], sm[NN], red[128];
  int bi = blockIdx.x;            // b*N + i
  int b  = bi >> 8;
  int t  = threadIdx.x;           // 0..127 (k index in phase 2)
  const float* drow = dist + (size_t)bi * NN;
  const int*   arow = adj  + (size_t)bi * NN;
  float wa_d = Wa[2 * DD];
  float bav  = ba[0];
  float ai   = Ai[bi];
  int jbase = b * NN;

  // phase 1: logits (leaky-relu 0.2, mask -> -1e9)
  for (int j = t; j < NN; j += 128) {
    float d = drow[j];
    float e = ai + Aj[jbase + j] + d * wa_d + bav;
    e = (e > 0.f) ? e : 0.2f * e;
    float mk = (arow[j] != 0) ? 1.f : 0.f;
    if (mk == 0.f) e = -1e9f;
    se[j] = e; sd[j] = d; sm[j] = mk;
  }
  __syncthreads();

  // softmax: max then sum over all 256 j
  red[t] = fmaxf(se[t], se[t + 128]);
  __syncthreads();
  for (int s = 64; s > 0; s >>= 1) {
    if (t < s) red[t] = fmaxf(red[t], red[t + s]);
    __syncthreads();
  }
  float mx = red[0];
  __syncthreads();
  float e0 = __expf(se[t] - mx);
  float e1 = __expf(se[t + 128] - mx);
  red[t] = e0 + e1;
  __syncthreads();
  for (int s = 64; s > 0; s >>= 1) {
    if (t < s) red[t] += red[t + s];
    __syncthreads();
  }
  float inv = 1.f / red[0];
  sw[t]       = e0 * sm[t] * inv;        // softmax prob * mask
  sw[t + 128] = e1 * sm[t + 128] * inv;
  __syncthreads();

  // phase 2: weighted relu reduction over j (w uniform across block ->
  // uniform branch skips masked neighbors; HB row read is coalesced & L2-hot)
  float hak = HA[(size_t)bi * OUTD + t];
  float w1k = Wm1[(size_t)2 * DD * OUTD + t];  // Wm1 row 256
  float b1k = bm1[t];
  const float* HBb = HB + (size_t)jbase * OUTD;
  float acc = 0.f, wsum = 0.f;
  for (int j = 0; j < NN; ++j) {
    float w = sw[j];
    if (w == 0.f) continue;
    wsum += w;
    float pre = hak + HBb[(size_t)j * OUTD + t] + sd[j] * w1k + b1k;
    acc += w * fmaxf(pre, 0.f);
  }
  S[(size_t)bi * OUTD + t] = acc;
  if (t == 0) Wsum[bi] = wsum;
}

// ---------------------------------------------------------------------------
// Kernel 5: h_new = relu( h@Wu[0:128] + S@Wfused + wsum (x) cvec + bu )
// WMMA GEMM, grid (BN/16, OUT/16), block 32.
// ---------------------------------------------------------------------------
__global__ void k_output(const float* __restrict__ h,
                         const float* __restrict__ S,
                         const float* __restrict__ Wu,
                         const float* __restrict__ Wfused,
                         const float* __restrict__ Wsum,
                         const float* __restrict__ cvec,
                         const float* __restrict__ bu,
                         float* __restrict__ out) {
  int lane  = threadIdx.x;
  int mBase = blockIdx.x * 16;
  int nBase = blockIdx.y * 16;
  v8f acc = {};
#pragma unroll
  for (int k = 0; k < DD; k += 32) {
    v16h a = load_frag_a(h, DD, mBase, k, lane);
    v16h b = load_frag_b(Wu, OUTD, k, nBase, lane);
    acc = __builtin_amdgcn_wmma_f32_16x16x32_f16(false, a, false, b,
                                                 (short)0, acc, false, false);
  }
#pragma unroll
  for (int k = 0; k < OUTD; k += 32) {
    v16h a = load_frag_a(S, OUTD, mBase, k, lane);
    v16h b = load_frag_b(Wfused, OUTD, k, nBase, lane);
    acc = __builtin_amdgcn_wmma_f32_16x16x32_f16(false, a, false, b,
                                                 (short)0, acc, false, false);
  }
  int n = nBase + (lane & 15);
  int mAdd = (lane >> 4) * 8;
  float cn = cvec[n];
  float bn = bu[n];
#pragma unroll
  for (int r = 0; r < 8; ++r) {
    int row = mBase + r + mAdd;
    float v = acc[r] + Wsum[row] * cn + bn;
    out[(size_t)row * OUTD + n] = fmaxf(v, 0.f);
  }
}

// ---------------------------------------------------------------------------
extern "C" void kernel_launch(void* const* d_in, const int* in_sizes, int n_in,
                              void* d_out, int out_size, void* d_ws, size_t ws_size,
                              hipStream_t stream) {
  (void)in_sizes; (void)n_in; (void)out_size; (void)ws_size;
  const float* h    = (const float*)d_in[0];
  const int*   adj  = (const int*)  d_in[1];
  const float* dist = (const float*)d_in[2];
  const float* Wm1  = (const float*)d_in[3];
  const float* bm1  = (const float*)d_in[4];
  const float* Wm2  = (const float*)d_in[5];
  const float* bm2  = (const float*)d_in[6];
  const float* Wa   = (const float*)d_in[7];
  const float* ba   = (const float*)d_in[8];
  const float* Wu   = (const float*)d_in[9];
  const float* bu   = (const float*)d_in[10];
  float* out = (float*)d_out;

  // workspace layout (bytes); total ~3.24 MB
  char* ws = (char*)d_ws;
  float* HA     = (float*)(ws + 0);                 // BN*OUT f32 = 1 MB
  float* HB     = (float*)(ws + 1048576);           // 1 MB
  float* Ai     = (float*)(ws + 2097152);           // 8 KB
  float* Aj     = (float*)(ws + 2105344);           // 8 KB
  float* S      = (float*)(ws + 2113536);           // 1 MB
  float* Wsum   = (float*)(ws + 3162112);           // 8 KB
  float* Wfused = (float*)(ws + 3170304);           // 64 KB
  float* cvec   = (float*)(ws + 3235840);           // 512 B

  k_gemm_h_wm1 <<<dim3(BN / 16, OUTD / 16, 2), 32, 0, stream>>>(h, Wm1, HA, HB);
  k_attn_scalars<<<BN, DD, 0, stream>>>(h, Wa, Ai, Aj);
  k_fuse       <<<OUTD + 1, OUTD, 0, stream>>>(Wm2, Wu, bm2, Wfused, cvec);
  k_softmax_agg<<<BN, 128, 0, stream>>>(dist, adj, Ai, Aj, HA, HB,
                                        Wm1, bm1, Wa, ba, S, Wsum);
  k_output     <<<dim3(BN / 16, OUTD / 16), 32, 0, stream>>>(h, S, Wu, Wfused,
                                                             Wsum, cvec, bu, out);
}
